// GCNBayesian_7610682049034
// MI455X (gfx1250) — matrix operand order repaired and checked by reading, not compiled, not run on hardware
//
#include <hip/hip_runtime.h>
#include <hip/hip_bf16.h>

typedef __attribute__((ext_vector_type(16))) __bf16 v16bf;
typedef __attribute__((ext_vector_type(8)))  __bf16 v8bf;
typedef __attribute__((ext_vector_type(8)))  float  v8f;

union BF16x16 { v16bf v; v8bf h[2]; };

// ---------------------------------------------------------------------------
// Weight transpose + f32->bf16 convert: W[K][Nout] -> WT[Nout][K] (bf16)
// ---------------------------------------------------------------------------
__global__ void k_prep_w(const float* __restrict__ W, __bf16* __restrict__ WT,
                         int K, int Nout) {
    int idx = blockIdx.x * blockDim.x + threadIdx.x;
    int total = K * Nout;
    if (idx >= total) return;
    int k = idx / Nout;
    int n = idx - k * Nout;
    WT[(size_t)n * K + k] = (__bf16)W[idx];
}

// ---------------------------------------------------------------------------
// f32 -> bf16 convert
// ---------------------------------------------------------------------------
__global__ void k_f2bf(const float* __restrict__ in, __bf16* __restrict__ out,
                       long long n) {
    long long idx = (long long)blockIdx.x * blockDim.x + threadIdx.x;
    if (idx >= n) return;
    out[idx] = (__bf16)in[idx];
}

// ---------------------------------------------------------------------------
// Degree: init to 1.0 (self loop), accumulate edge weights, rsqrt
// ---------------------------------------------------------------------------
__global__ void k_deg_init(float* __restrict__ deg, int N) {
    int i = blockIdx.x * blockDim.x + threadIdx.x;
    if (i < N) deg[i] = 1.0f;
}

__global__ void k_deg_edges(const int* __restrict__ ei, const float* __restrict__ ew,
                            float* __restrict__ deg, int E) {
    int e = blockIdx.x * blockDim.x + threadIdx.x;
    if (e >= E) return;
    int d = ei[E + e];               // dst row
    atomicAdd(&deg[d], ew[e]);
}

__global__ void k_dinv(float* __restrict__ deg, int N) {
    int i = blockIdx.x * blockDim.x + threadIdx.x;
    if (i < N) deg[i] = rsqrtf(deg[i]);   // deg >= 1 always (self loop)
}

// ---------------------------------------------------------------------------
// WMMA GEMM: C[M x Nout] = A[M x K](bf16, row-major) * BT[Nout x K](bf16)^T
// One wave (32 threads) per 16x16 output tile.
// A lane layout (16-bit A 16x32): lane = half*16 + m; element j holds
//   K = k0 + (j>=8?16:0) + half*8 + (j%8)  -> two contiguous 8-bf16 chunks.
// B lane layout (16-bit B 32x16): lane = half*16 + n; element j holds
//   K = k0 + (j>=8?16:0) + half*8 + (j%8)  -> two contiguous 8-bf16 chunks.
// ---------------------------------------------------------------------------
__global__ __launch_bounds__(32)
void k_gemm_wmma(const __bf16* __restrict__ A, const __bf16* __restrict__ BT,
                 float* __restrict__ C, int K, int Nout) {
    const int lane = threadIdx.x & 31;
    const int half = lane >> 4;
    const int l16  = lane & 15;
    const int m0 = blockIdx.x << 4;
    const int n0 = blockIdx.y << 4;

    const __bf16* arow = A  + (size_t)(m0 + l16) * K;
    const __bf16* brow = BT + (size_t)(n0 + l16) * K;

    v8f acc = {};
    for (int k0 = 0; k0 < K; k0 += 32) {
        BF16x16 a, b;
        a.h[0] = *(const v8bf*)(arow + k0 + half * 8);
        a.h[1] = *(const v8bf*)(arow + k0 + 16 + half * 8);
        b.h[0] = *(const v8bf*)(brow + k0 + half * 8);
        b.h[1] = *(const v8bf*)(brow + k0 + 16 + half * 8);
        acc = __builtin_amdgcn_wmma_f32_16x16x32_bf16(
            /*neg_a=*/false, a.v, /*neg_b=*/false, b.v,
            /*c_mod=*/(short)0, acc, /*reuse_a=*/false, /*reuse_b=*/false);
    }

    // C/D layout: lane L, VGPR v -> (M = m0 + half*8 + v, N = n0 + L%16)
    float* crow = C + (size_t)(m0 + half * 8) * Nout + n0 + l16;
    #pragma unroll
    for (int v = 0; v < 8; ++v)
        crow[(size_t)v * Nout] = acc[v];
}

// ---------------------------------------------------------------------------
// agg[i][f] = bias[f] + hw[i][f] * dinv[i]^2   (self-loop term + bias)
// ---------------------------------------------------------------------------
__global__ void k_init_agg(const float* __restrict__ hw, const float* __restrict__ dinv,
                           const float* __restrict__ bias, float* __restrict__ agg,
                           int total, int fshift) {
    int idx = blockIdx.x * blockDim.x + threadIdx.x;
    if (idx >= total) return;
    int i = idx >> fshift;
    int f = idx & ((1 << fshift) - 1);
    float di = dinv[i];
    agg[idx] = bias[f] + hw[idx] * di * di;
}

// ---------------------------------------------------------------------------
// Edge scatter: agg[dst] += hw[src] * (dinv[src]*ew*dinv[dst])
// One thread per (edge, 4-feature chunk); float4 gather + 4 global atomics.
// ---------------------------------------------------------------------------
__global__ void k_scatter(const float* __restrict__ hw, const int* __restrict__ ei,
                          const float* __restrict__ ew, const float* __restrict__ dinv,
                          float* __restrict__ agg, int E, int F, int cshift) {
    int t = blockIdx.x * blockDim.x + threadIdx.x;
    int chunks = 1 << cshift;
    if (t >= (E << cshift)) return;
    int e = t >> cshift;
    int c = t & (chunks - 1);
    int s = ei[e];
    int d = ei[E + e];
    float nrm = dinv[s] * ew[e] * dinv[d];
    const float* srcp = hw + (size_t)s * F + c * 4;
    __builtin_prefetch(srcp, 0, 0);
    float4 v = *(const float4*)srcp;
    float* dstp = agg + (size_t)d * F + c * 4;
    atomicAdd(dstp + 0, v.x * nrm);
    atomicAdd(dstp + 1, v.y * nrm);
    atomicAdd(dstp + 2, v.z * nrm);
    atomicAdd(dstp + 3, v.w * nrm);
}

// ---------------------------------------------------------------------------
// ELU + convert to bf16 (and optional f32 copy for the pooling stage)
// ---------------------------------------------------------------------------
__global__ void k_elu(const float* __restrict__ agg, __bf16* __restrict__ outb,
                      float* __restrict__ outf, int total) {
    int idx = blockIdx.x * blockDim.x + threadIdx.x;
    if (idx >= total) return;
    float v = agg[idx];
    float h = v > 0.0f ? v : expm1f(v);
    outb[idx] = (__bf16)h;
    if (outf) outf[idx] = h;
}

// ---------------------------------------------------------------------------
// Zero small buffer
// ---------------------------------------------------------------------------
__global__ void k_zero(float* __restrict__ p, int n) {
    int i = blockIdx.x * blockDim.x + threadIdx.x;
    if (i < n) p[i] = 0.0f;
}

// ---------------------------------------------------------------------------
// Global mean pool (stage 1): LDS-accumulated per-block partial sums of
// h[N x 32] by graph id, then flushed with global atomics. B = 64 graphs.
// ---------------------------------------------------------------------------
__global__ __launch_bounds__(256)
void k_pool(const float* __restrict__ h, const int* __restrict__ batch,
            float* __restrict__ pool, float* __restrict__ cnt, int N) {
    __shared__ float sp[64 * 32];
    __shared__ float sc[64];
    for (int i = threadIdx.x; i < 64 * 32; i += blockDim.x) sp[i] = 0.0f;
    for (int i = threadIdx.x; i < 64; i += blockDim.x) sc[i] = 0.0f;
    __syncthreads();

    int node = blockIdx.x * blockDim.x + threadIdx.x;
    if (node < N) {
        int b = batch[node];
        atomicAdd(&sc[b], 1.0f);
        const float* row = h + (size_t)node * 32;
        #pragma unroll
        for (int f = 0; f < 32; ++f)
            atomicAdd(&sp[b * 32 + f], row[f]);
    }
    __syncthreads();

    for (int i = threadIdx.x; i < 64 * 32; i += blockDim.x)
        if (sp[i] != 0.0f) atomicAdd(&pool[i], sp[i]);
    for (int i = threadIdx.x; i < 64; i += blockDim.x)
        if (sc[i] != 0.0f) atomicAdd(&cnt[i], sc[i]);
}

// ---------------------------------------------------------------------------
// Head: mean pool finish, LayerNorm(32), FC(32->8)+ELU, Bayesian linear.
// One block of 64 threads; thread b handles graph b.
// ---------------------------------------------------------------------------
__global__ __launch_bounds__(64)
void k_head(const float* __restrict__ pool, const float* __restrict__ cnt,
            const float* __restrict__ ln_g, const float* __restrict__ ln_b,
            const float* __restrict__ fc_w, const float* __restrict__ fc_b,
            const float* __restrict__ w_mu, const float* __restrict__ w_rho,
            const float* __restrict__ w_eps, const float* __restrict__ b_mu,
            const float* __restrict__ b_rho, const float* __restrict__ b_eps,
            float* __restrict__ out) {
    int b = threadIdx.x;
    if (b >= 64) return;
    float c = fmaxf(cnt[b], 1.0f);
    float g[32];
    float mu = 0.0f;
    #pragma unroll
    for (int f = 0; f < 32; ++f) { g[f] = pool[b * 32 + f] / c; mu += g[f]; }
    mu *= (1.0f / 32.0f);
    float var = 0.0f;
    #pragma unroll
    for (int f = 0; f < 32; ++f) { float d = g[f] - mu; var += d * d; }
    var *= (1.0f / 32.0f);
    float inv = rsqrtf(var + 1e-5f);
    #pragma unroll
    for (int f = 0; f < 32; ++f)
        g[f] = (g[f] - mu) * inv * ln_g[f] + ln_b[f];

    float o = b_mu[0] + log1pf(expf(b_rho[0])) * b_eps[0];
    #pragma unroll
    for (int j = 0; j < 8; ++j) {
        float s = fc_b[j];
        #pragma unroll
        for (int f = 0; f < 32; ++f) s += g[f] * fc_w[f * 8 + j];
        float h2 = s > 0.0f ? s : expm1f(s);
        float w = w_mu[j] + log1pf(expf(w_rho[j])) * w_eps[j];
        o += h2 * w;
    }
    out[b] = o;   // [B,1]
}

// ---------------------------------------------------------------------------
extern "C" void kernel_launch(void* const* d_in, const int* in_sizes, int n_in,
                              void* d_out, int out_size, void* d_ws, size_t ws_size,
                              hipStream_t stream) {
    const float* x     = (const float*)d_in[0];
    const int*   ei    = (const int*)d_in[1];
    const int*   batch = (const int*)d_in[2];
    const float* ew    = (const float*)d_in[3];
    const float* W1    = (const float*)d_in[4];
    const float* b1    = (const float*)d_in[5];
    const float* W2    = (const float*)d_in[6];
    const float* b2    = (const float*)d_in[7];
    const float* W3    = (const float*)d_in[8];
    const float* b3    = (const float*)d_in[9];
    const float* ln_g  = (const float*)d_in[10];
    const float* ln_b  = (const float*)d_in[11];
    const float* fc_w  = (const float*)d_in[12];
    const float* fc_b  = (const float*)d_in[13];
    const float* w_mu  = (const float*)d_in[14];
    const float* w_rho = (const float*)d_in[15];
    const float* w_eps = (const float*)d_in[16];
    const float* b_mu  = (const float*)d_in[17];
    const float* b_rho = (const float*)d_in[18];
    const float* b_eps = (const float*)d_in[19];
    float* out = (float*)d_out;

    const int N = in_sizes[2];        // 100000
    const int E = in_sizes[3];        // 640000

    // ---- workspace layout (256B aligned) ----
    char* base = (char*)d_ws;
    size_t off = 0;
    auto alloc = [&](size_t bytes) -> void* {
        void* p = base + off;
        off = (off + bytes + 255) & ~(size_t)255;
        return p;
    };
    float*  dinv = (float*)alloc((size_t)N * 4);
    __bf16* xb   = (__bf16*)alloc((size_t)N * 128 * 2);
    __bf16* hb   = (__bf16*)alloc((size_t)N * 128 * 2);
    float*  hw   = (float*)alloc((size_t)N * 128 * 4);
    float*  agg  = (float*)alloc((size_t)N * 128 * 4);
    __bf16* w1t  = (__bf16*)alloc(128 * 128 * 2);
    __bf16* w2t  = (__bf16*)alloc(64 * 128 * 2);
    __bf16* w3t  = (__bf16*)alloc(32 * 64 * 2);
    float*  pool = (float*)alloc(64 * 32 * 4);
    float*  cnt  = (float*)alloc(64 * 4);
    (void)ws_size;

    const int TB = 256;
    auto blks = [&](long long n) { return (unsigned)((n + TB - 1) / TB); };

    // weights -> bf16 transposed
    k_prep_w<<<blks(128 * 128), TB, 0, stream>>>(W1, w1t, 128, 128);
    k_prep_w<<<blks(128 * 64),  TB, 0, stream>>>(W2, w2t, 128, 64);
    k_prep_w<<<blks(64 * 32),   TB, 0, stream>>>(W3, w3t, 64, 32);
    // x -> bf16
    k_f2bf<<<blks((long long)N * 128), TB, 0, stream>>>(x, xb, (long long)N * 128);

    // gcn_norm: dinv = rsqrt(deg) with self loops
    k_deg_init <<<blks(N), TB, 0, stream>>>(dinv, N);
    k_deg_edges<<<blks(E), TB, 0, stream>>>(ei, ew, dinv, E);
    k_dinv     <<<blks(N), TB, 0, stream>>>(dinv, N);

    const int Mt = N / 16;            // 6250, exact

    // ---- layer 1: 128 -> 128 ----
    k_gemm_wmma<<<dim3(Mt, 128 / 16), 32, 0, stream>>>(xb, w1t, hw, 128, 128);
    k_init_agg<<<blks((long long)N * 128), TB, 0, stream>>>(hw, dinv, b1, agg, N * 128, 7);
    k_scatter<<<blks((long long)E << 5), TB, 0, stream>>>(hw, ei, ew, dinv, agg, E, 128, 5);
    k_elu<<<blks((long long)N * 128), TB, 0, stream>>>(agg, hb, nullptr, N * 128);

    // ---- layer 2: 128 -> 64 ----
    k_gemm_wmma<<<dim3(Mt, 64 / 16), 32, 0, stream>>>(hb, w2t, hw, 128, 64);
    k_init_agg<<<blks((long long)N * 64), TB, 0, stream>>>(hw, dinv, b2, agg, N * 64, 6);
    k_scatter<<<blks((long long)E << 4), TB, 0, stream>>>(hw, ei, ew, dinv, agg, E, 64, 4);
    k_elu<<<blks((long long)N * 64), TB, 0, stream>>>(agg, hb, nullptr, N * 64);

    // ---- layer 3: 64 -> 32 ----
    k_gemm_wmma<<<dim3(Mt, 32 / 16), 32, 0, stream>>>(hb, w3t, hw, 64, 32);
    k_init_agg<<<blks((long long)N * 32), TB, 0, stream>>>(hw, dinv, b3, agg, N * 32, 5);
    k_scatter<<<blks((long long)E << 3), TB, 0, stream>>>(hw, ei, ew, dinv, agg, E, 32, 3);
    // ELU result as f32 into hw (hw no longer needed) for pooling
    k_elu<<<blks((long long)N * 32), TB, 0, stream>>>(agg, hb, hw, N * 32);

    // ---- pooling + head ----
    k_zero<<<blks(64 * 32), TB, 0, stream>>>(pool, 64 * 32);
    k_zero<<<1, 64, 0, stream>>>(cnt, 64);
    k_pool<<<blks(N), TB, 0, stream>>>(hw, batch, pool, cnt, N);
    k_head<<<1, 64, 0, stream>>>(pool, cnt, ln_g, ln_b, fc_w, fc_b,
                                 w_mu, w_rho, w_eps, b_mu, b_rho, b_eps, out);
}
